// NeuralTranslator_52338471469448
// MI455X (gfx1250) — matrix operand order repaired and checked by reading, not compiled, not run on hardware
//
#include <hip/hip_runtime.h>
#include <math.h>

// ---------------- problem constants ----------------
constexpr int kB = 32;      // batch
constexpr int kS = 64;      // source length
constexpr int kL = 64;      // target length
constexpr int kE = 512;     // embed dim
constexpr int kH = 1024;    // hidden
constexpr int kV = 32000;   // vocab out

// ---------------- vector types for WMMA ----------------
typedef __attribute__((ext_vector_type(16))) __bf16 v16bf;
typedef __attribute__((ext_vector_type(8)))  __bf16 v8bf;
typedef __attribute__((ext_vector_type(4)))  __bf16 v4bf;
typedef __attribute__((ext_vector_type(8)))  float  v8f;
typedef __attribute__((ext_vector_type(4)))  float  vf4;

__device__ __forceinline__ float sigf(float x) { return 1.0f / (1.0f + __expf(-x)); }

// =====================================================================
// f32 -> bf16 bulk conversion (4 elems / thread); all sizes % 4 == 0
// =====================================================================
__global__ void k_cvt4(const float* __restrict__ in, __bf16* __restrict__ out, int n4)
{
    int i = blockIdx.x * blockDim.x + threadIdx.x;
    if (i >= n4) return;
    vf4 v = ((const vf4*)in)[i];
    v4bf o;
    o[0] = (__bf16)v[0]; o[1] = (__bf16)v[1];
    o[2] = (__bf16)v[2]; o[3] = (__bf16)v[3];
    ((v4bf*)out)[i] = o;
}

// =====================================================================
// Embedding + sinusoidal positional encoding, output bf16.
// =====================================================================
__global__ void k_embed(const int* __restrict__ tok, int tok_ld,
                        const float* __restrict__ emb,
                        __bf16* __restrict__ out,
                        int rows, int steps, int use_pos)
{
    int i = blockIdx.x * blockDim.x + threadIdx.x;
    int total = rows * steps * kE;
    if (i >= total) return;
    int e = i % kE;
    int rs = i / kE;
    int s = rs % steps;
    int b = rs / steps;
    int t = tok[b * tok_ld + s];
    float x = emb[(size_t)t * kE + e];
    int half2 = (e >> 1) * 2;
    float div = __expf((float)half2 * (-logf(10000.0f) / (float)kE));
    float pos = use_pos ? (float)s : 0.0f;
    float pe = (e & 1) ? __cosf(pos * div) : __sinf(pos * div);
    out[i] = (__bf16)(x + pe);
}

// =====================================================================
// WMMA GEMM, all-bf16 operands:  C[m,n] = sum_k A[m,k]*Wb[n,k] (+bias[n])
// Wave tile 32(M) x 64(N): 8 WMMAs per k-step, 12 x 16B loads.
// Block = 4 waves -> 32 x 256 tile; guards wave-uniform (EXEC all ones).
// out_mode 0: C[m*ldc + n]          (regular temporal)
// out_mode 1: m = t*32 + b  ->  C[b*kL*kV + (t+1)*kV + n], NON-TEMPORAL
//             stores so the streamed 262 MB output doesn't evict W2
//             from L2 between row-strip sweeps.
// =====================================================================
__global__ void __launch_bounds__(128)
k_gemm(const __bf16* __restrict__ A, const __bf16* __restrict__ Wb,
       const float* __restrict__ bias, float* __restrict__ C,
       int M, int N, int K, int lda, int ldw, int ldc, int out_mode)
{
    const int lane = threadIdx.x & 31;
    const int wave = threadIdx.x >> 5;
    const int row0 = blockIdx.y << 5;            // 32-row strip
    const int n0   = blockIdx.x * 256 + wave * 64;
    if (row0 >= M || n0 >= N) return;            // wave-uniform

    const int halfsel = lane >> 4;               // 0 | 1
    const int kbandA  = halfsel * 8;
    const int kbandB  = halfsel * 8;
    const int rowA0   = row0 + (lane & 15);
    const int rowA1   = rowA0 + 16;
    const int colB    = lane & 15;

    v8f acc[2][4] = {};

    for (int k0 = 0; k0 < K; k0 += 32) {
        const __bf16* ap0 = A + (size_t)rowA0 * lda + k0 + kbandA;
        const __bf16* ap1 = A + (size_t)rowA1 * lda + k0 + kbandA;
        union { v16bf v; v8bf h2[2]; } ua0, ua1;
        ua0.h2[0] = *(const v8bf*)(ap0);
        ua0.h2[1] = *(const v8bf*)(ap0 + 16);
        ua1.h2[0] = *(const v8bf*)(ap1);
        ua1.h2[1] = *(const v8bf*)(ap1 + 16);

        #pragma unroll
        for (int j = 0; j < 4; ++j) {
            const __bf16* wpb = Wb + (size_t)(n0 + j * 16 + colB) * ldw + k0 + kbandB;
            union { v16bf v; v8bf h2[2]; } ub;
            ub.h2[0] = *(const v8bf*)(wpb);
            ub.h2[1] = *(const v8bf*)(wpb + 16);
            __builtin_prefetch((const void*)(wpb + 32), 0, 1);

            acc[0][j] = __builtin_amdgcn_wmma_f32_16x16x32_bf16(
                false, ua0.v, false, ub.v, (short)0, acc[0][j], false, false);
            acc[1][j] = __builtin_amdgcn_wmma_f32_16x16x32_bf16(
                false, ua1.v, false, ub.v, (short)0, acc[1][j], false, false);
        }
    }

    #pragma unroll
    for (int r = 0; r < 2; ++r) {
        const int mbase = row0 + r * 16 + halfsel * 8;
        #pragma unroll
        for (int j = 0; j < 4; ++j) {
            int n = n0 + j * 16 + colB;
            float badd = bias ? bias[n] : 0.0f;
            #pragma unroll
            for (int v = 0; v < 8; ++v) {
                int m = mbase + v;
                float val = acc[r][j][v] + badd;
                if (out_mode == 0) {
                    C[(size_t)m * ldc + n] = val;
                } else {
                    int t = m >> 5;          // m = t*32 + b
                    int b = m & 31;
                    __builtin_nontemporal_store(
                        val, &C[(size_t)b * kL * kV + (size_t)(t + 1) * kV + n]);
                }
            }
        }
    }
}

// =====================================================================
// Fused recurrent step: gates = h_src @ Whh^T + pre[t] + bih + bhh, then
// LSTM pointwise. 16 blocks x 128 threads; block owns H-slice j0..j0+63.
// Wave w (0..3) computes gate quadrant w for that slice (n = w*H + j0 +..)
// -> 32x64 tile each, staged in LDS, then 128 threads apply nonlinearity.
// h double-buffered across steps (block reads ALL of h_src but writes
// only its own slice of h_dst / c).  c slice is block-private.
// =====================================================================
__global__ void __launch_bounds__(128)
k_step_fused(const __bf16* __restrict__ hsrc, __bf16* __restrict__ hdst,
             const __bf16* __restrict__ Whh_b,
             const float* __restrict__ pre, int pre_row_stride, int pre_off,
             const float* __restrict__ bih, const float* __restrict__ bhh,
             float* __restrict__ c)
{
    __shared__ float gsh[4][32][64];             // 32 KB: i,f,g,o tiles
    const int lane = threadIdx.x & 31;
    const int wave = threadIdx.x >> 5;           // gate quadrant (i,f,g,o)
    const int j0   = blockIdx.x * 64;            // H-slice
    const int nbase = wave * kH + j0;            // column base in 4H space

    const int halfsel = lane >> 4;
    const int kbandA  = halfsel * 8;
    const int kbandB  = halfsel * 8;
    const int rowA0   = lane & 15;
    const int rowA1   = rowA0 + 16;
    const int colB    = lane & 15;

    v8f acc[2][4] = {};

    for (int k0 = 0; k0 < kH; k0 += 32) {
        const __bf16* ap0 = hsrc + (size_t)rowA0 * kH + k0 + kbandA;
        const __bf16* ap1 = hsrc + (size_t)rowA1 * kH + k0 + kbandA;
        union { v16bf v; v8bf h2[2]; } ua0, ua1;
        ua0.h2[0] = *(const v8bf*)(ap0);
        ua0.h2[1] = *(const v8bf*)(ap0 + 16);
        ua1.h2[0] = *(const v8bf*)(ap1);
        ua1.h2[1] = *(const v8bf*)(ap1 + 16);

        #pragma unroll
        for (int j = 0; j < 4; ++j) {
            const __bf16* wpb = Whh_b + (size_t)(nbase + j * 16 + colB) * kH + k0 + kbandB;
            union { v16bf v; v8bf h2[2]; } ub;
            ub.h2[0] = *(const v8bf*)(wpb);
            ub.h2[1] = *(const v8bf*)(wpb + 16);

            acc[0][j] = __builtin_amdgcn_wmma_f32_16x16x32_bf16(
                false, ua0.v, false, ub.v, (short)0, acc[0][j], false, false);
            acc[1][j] = __builtin_amdgcn_wmma_f32_16x16x32_bf16(
                false, ua1.v, false, ub.v, (short)0, acc[1][j], false, false);
        }
    }

    // epilogue: + pre + biases, stage gate tile in LDS
    #pragma unroll
    for (int r = 0; r < 2; ++r) {
        const int mbase = r * 16 + halfsel * 8;
        #pragma unroll
        for (int j = 0; j < 4; ++j) {
            int col = j * 16 + colB;
            int n = nbase + col;
            float bsum = bih[n] + bhh[n];
            #pragma unroll
            for (int v = 0; v < 8; ++v) {
                int m = mbase + v;
                gsh[wave][m][col] = acc[r][j][v] +
                    pre[(size_t)m * pre_row_stride + pre_off + n] + bsum;
            }
        }
    }
    __syncthreads();

    // pointwise LSTM for this block's H-slice (2048 elems, 16/thread)
    #pragma unroll
    for (int e = 0; e < 16; ++e) {
        int flat = threadIdx.x + e * 128;        // 0..2047
        int b = flat >> 6;                       // 0..31
        int col = flat & 63;
        int j = j0 + col;
        float ig = gsh[0][b][col], fg = gsh[1][b][col];
        float gg = gsh[2][b][col], og = gsh[3][b][col];
        int ci = b * kH + j;
        float cn = sigf(fg) * c[ci] + sigf(ig) * tanhf(gg);
        float hn = sigf(og) * tanhf(cn);
        c[ci] = cn;
        hdst[ci] = (__bf16)hn;
    }
}

// =====================================================================
// Fused decoder head: z = h @ W1^T + b1 ; LayerNorm ; ELU ; bf16 out.
// One block, 256 threads = 8 waves; wave w computes the 32 x 128 strip
// n in [w*128, w*128+128).  z staged via global ztmp (intra-block fence),
// then per-row LN with 8 threads/row.
// =====================================================================
__global__ void __launch_bounds__(256)
k_zproj_ln(const __bf16* __restrict__ hb, const __bf16* __restrict__ W1b,
           const float* __restrict__ b1, const float* __restrict__ g,
           const float* __restrict__ be, float* __restrict__ ztmp,
           __bf16* __restrict__ zout)
{
    const int lane = threadIdx.x & 31;
    const int wave = threadIdx.x >> 5;           // 0..7
    const int n0   = wave * 128;

    const int halfsel = lane >> 4;
    const int kbandA  = halfsel * 8;
    const int kbandB  = halfsel * 8;
    const int rowA0   = lane & 15;
    const int rowA1   = rowA0 + 16;
    const int colB    = lane & 15;

    v8f acc[2][8] = {};

    for (int k0 = 0; k0 < kH; k0 += 32) {
        const __bf16* ap0 = hb + (size_t)rowA0 * kH + k0 + kbandA;
        const __bf16* ap1 = hb + (size_t)rowA1 * kH + k0 + kbandA;
        union { v16bf v; v8bf h2[2]; } ua0, ua1;
        ua0.h2[0] = *(const v8bf*)(ap0);
        ua0.h2[1] = *(const v8bf*)(ap0 + 16);
        ua1.h2[0] = *(const v8bf*)(ap1);
        ua1.h2[1] = *(const v8bf*)(ap1 + 16);

        #pragma unroll
        for (int j = 0; j < 8; ++j) {
            const __bf16* wpb = W1b + (size_t)(n0 + j * 16 + colB) * kH + k0 + kbandB;
            union { v16bf v; v8bf h2[2]; } ub;
            ub.h2[0] = *(const v8bf*)(wpb);
            ub.h2[1] = *(const v8bf*)(wpb + 16);

            acc[0][j] = __builtin_amdgcn_wmma_f32_16x16x32_bf16(
                false, ua0.v, false, ub.v, (short)0, acc[0][j], false, false);
            acc[1][j] = __builtin_amdgcn_wmma_f32_16x16x32_bf16(
                false, ua1.v, false, ub.v, (short)0, acc[1][j], false, false);
        }
    }

    // z (+ b1) -> ztmp
    #pragma unroll
    for (int r = 0; r < 2; ++r) {
        const int mbase = r * 16 + halfsel * 8;
        #pragma unroll
        for (int j = 0; j < 8; ++j) {
            int n = n0 + j * 16 + colB;
            float badd = b1[n];
            #pragma unroll
            for (int v = 0; v < 8; ++v)
                ztmp[(size_t)(mbase + v) * kH + n] = acc[r][j][v] + badd;
        }
    }
    __threadfence_block();
    __syncthreads();

    // LayerNorm + ELU: 8 threads per row
    __shared__ float rs1[32][8], rs2[32][8], rmu[32], rinv[32];
    const int row = threadIdx.x >> 3;            // 0..31
    const int sub = threadIdx.x & 7;             // 0..7
    float sum = 0.0f, sq = 0.0f;
    for (int q = 0; q < 128; ++q) {
        float x = ztmp[(size_t)row * kH + sub * 128 + q];
        sum += x; sq += x * x;
    }
    rs1[row][sub] = sum; rs2[row][sub] = sq;
    __syncthreads();
    if (sub == 0) {
        float s = 0.0f, ss = 0.0f;
        #pragma unroll
        for (int q = 0; q < 8; ++q) { s += rs1[row][q]; ss += rs2[row][q]; }
        float mu = s * (1.0f / kH);
        float var = ss * (1.0f / kH) - mu * mu;
        rmu[row] = mu;
        rinv[row] = rsqrtf(var + 1e-5f);
    }
    __syncthreads();
    float mu = rmu[row], inv = rinv[row];
    for (int q = 0; q < 128; ++q) {
        int j = sub * 128 + q;
        float x = ztmp[(size_t)row * kH + j];
        float y = (x - mu) * inv * g[j] + be[j];
        y = (y > 0.0f) ? y : (__expf(y) - 1.0f);
        zout[(size_t)row * kH + j] = (__bf16)y;
    }
}

// ---------------- zero helpers ----------------
__global__ void k_zero_f32(float* p, int n)
{ int i = blockIdx.x * blockDim.x + threadIdx.x; if (i < n) p[i] = 0.0f; }

__global__ void k_zero_bf16(__bf16* p, int n)
{ int i = blockIdx.x * blockDim.x + threadIdx.x; if (i < n) p[i] = (__bf16)0.0f; }

__global__ void k_zero_out0(float* __restrict__ out)
{
    int i = blockIdx.x * blockDim.x + threadIdx.x;
    if (i >= kB * kV) return;
    int b = i / kV, v = i % kV;
    out[(size_t)b * kL * kV + v] = 0.0f;   // outputs[:, 0, :] = 0
}

// =====================================================================
// driver
// =====================================================================
extern "C" void kernel_launch(void* const* d_in, const int* in_sizes, int n_in,
                              void* d_out, int out_size, void* d_ws, size_t ws_size,
                              hipStream_t stream)
{
    (void)in_sizes; (void)n_in; (void)out_size; (void)ws_size;

    const int*   source  = (const int*)  d_in[0];
    const int*   target  = (const int*)  d_in[1];
    const float* enc_emb = (const float*)d_in[2];
    const float* enc_Wih = (const float*)d_in[3];
    const float* enc_Whh = (const float*)d_in[4];
    const float* enc_bih = (const float*)d_in[5];
    const float* enc_bhh = (const float*)d_in[6];
    const float* dec_emb = (const float*)d_in[7];
    const float* dec_Wih = (const float*)d_in[8];
    const float* dec_Whh = (const float*)d_in[9];
    const float* dec_bih = (const float*)d_in[10];
    const float* dec_bhh = (const float*)d_in[11];
    const float* W1      = (const float*)d_in[12];
    const float* b1      = (const float*)d_in[13];
    const float* ln_g    = (const float*)d_in[14];
    const float* ln_b    = (const float*)d_in[15];
    const float* W2      = (const float*)d_in[16];
    const float* b2      = (const float*)d_in[17];
    float* out = (float*)d_out;

    // ---- workspace carve-up (256B aligned) ----
    char* wp = (char*)d_ws;
    auto take = [&](size_t bytes) -> void* {
        void* r = (void*)wp;
        wp += (bytes + 255) & ~(size_t)255;
        return r;
    };
    float*  preEnc  = (float*) take(sizeof(float)  * (size_t)kB * kS * 4 * kH);       // 32 MB
    float*  preDec  = (float*) take(sizeof(float)  * (size_t)kB * (kL - 1) * 4 * kH); // 31.5 MB
    __bf16* xEnc    = (__bf16*)take(sizeof(__bf16) * (size_t)kB * kS * kE);
    __bf16* xDec    = (__bf16*)take(sizeof(__bf16) * (size_t)kB * (kL - 1) * kE);
    __bf16* zall    = (__bf16*)take(sizeof(__bf16) * (size_t)(kL - 1) * kB * kH);     // 4 MB
    __bf16* eWih_b  = (__bf16*)take(sizeof(__bf16) * (size_t)4 * kH * kE);
    __bf16* dWih_b  = (__bf16*)take(sizeof(__bf16) * (size_t)4 * kH * kE);
    __bf16* eWhh_b  = (__bf16*)take(sizeof(__bf16) * (size_t)4 * kH * kH);
    __bf16* dWhh_b  = (__bf16*)take(sizeof(__bf16) * (size_t)4 * kH * kH);
    __bf16* W1_b    = (__bf16*)take(sizeof(__bf16) * (size_t)kH * kH);
    __bf16* W2_b    = (__bf16*)take(sizeof(__bf16) * (size_t)kV * kH);                // 62.5 MB
    float*  c       = (float*) take(sizeof(float)  * (size_t)kB * kH);
    __bf16* hb0     = (__bf16*)take(sizeof(__bf16) * (size_t)kB * kH);
    __bf16* hb1     = (__bf16*)take(sizeof(__bf16) * (size_t)kB * kH);
    float*  ztmp    = (float*) take(sizeof(float)  * (size_t)kB * kH);

    auto cvt = [&](const float* src, __bf16* dst, int n) {
        int n4 = n / 4;
        k_cvt4<<<(n4 + 255) / 256, 256, 0, stream>>>(src, dst, n4);
    };
    auto gemm = [&](const __bf16* A, const __bf16* W, const float* bias, float* C,
                    int M, int N, int K, int lda, int ldw, int ldc, int mode) {
        dim3 g((N + 255) / 256, (M + 31) / 32);
        k_gemm<<<g, dim3(128), 0, stream>>>(A, W, bias, C, M, N, K, lda, ldw, ldc, mode);
    };

    // ---- one-time weight conversions to bf16 ----
    cvt(enc_Wih, eWih_b, 4 * kH * kE);
    cvt(dec_Wih, dWih_b, 4 * kH * kE);
    cvt(enc_Whh, eWhh_b, 4 * kH * kH);
    cvt(dec_Whh, dWhh_b, 4 * kH * kH);
    cvt(W1,      W1_b,   kH * kH);
    cvt(W2,      W2_b,   kV * kH);

    // ---- embeddings + PE (bf16) ----
    {
        int tot = kB * kS * kE;
        k_embed<<<(tot + 255) / 256, 256, 0, stream>>>(source, kS, enc_emb, xEnc, kB, kS, 1);
    }
    {
        int tot = kB * (kL - 1) * kE;   // target[:, :-1], pe position 0
        k_embed<<<(tot + 255) / 256, 256, 0, stream>>>(target, kL, dec_emb, xDec, kB, kL - 1, 0);
    }

    // ---- input-gate pre-GEMMs hoisted out of the scans ----
    gemm(xEnc, eWih_b, nullptr, preEnc, kB * kS,       4 * kH, kE, kE, kE, 4 * kH, 0);
    gemm(xDec, dWih_b, nullptr, preDec, kB * (kL - 1), 4 * kH, kE, kE, kE, 4 * kH, 0);

    // ---- zero initial state ----
    k_zero_f32 <<<(kB * kH + 255) / 256, 256, 0, stream>>>(c,   kB * kH);
    k_zero_bf16<<<(kB * kH + 255) / 256, 256, 0, stream>>>(hb0, kB * kH);

    __bf16* hsrc = hb0;
    __bf16* hdst = hb1;

    // ---- encoder scan: 1 fused launch per step ----
    for (int t = 0; t < kS; ++t) {
        k_step_fused<<<16, 128, 0, stream>>>(
            hsrc, hdst, eWhh_b, preEnc, kS * 4 * kH, t * 4 * kH, enc_bih, enc_bhh, c);
        __bf16* tmp = hsrc; hsrc = hdst; hdst = tmp;
    }

    // ---- decoder scan: fused step + fused head per step ----
    for (int t = 0; t < kL - 1; ++t) {
        k_step_fused<<<16, 128, 0, stream>>>(
            hsrc, hdst, dWhh_b, preDec, (kL - 1) * 4 * kH, t * 4 * kH, dec_bih, dec_bhh, c);
        __bf16* tmp = hsrc; hsrc = hdst; hdst = tmp;
        k_zproj_ln<<<1, 256, 0, stream>>>(
            hsrc, W1_b, b1, ln_g, ln_b, ztmp, zall + (size_t)t * kB * kH);
    }

    // ---- outputs[:, 0, :] = 0 ----
    k_zero_out0<<<(kB * kV + 255) / 256, 256, 0, stream>>>(out);

    // ---- one batched output projection: [2016,1024] x [1024,32000] + b2 ----
    // W2 bf16 (65.5 MB) stays L2-resident; output stored non-temporally.
    gemm(zall, W2_b, b2, out, (kL - 1) * kB, kV, kH, kH, kH, 0, 1);
}
